// Attention_37795712204872
// MI455X (gfx1250) — matrix operand order repaired and checked by reading, not compiled, not run on hardware
//
#include <hip/hip_runtime.h>
#include <hip/hip_bf16.h>

typedef __attribute__((ext_vector_type(16))) __bf16 v16bf;
typedef __attribute__((ext_vector_type(8)))  __bf16 v8bf;
typedef __attribute__((ext_vector_type(4)))  __bf16 v4bf;
typedef __attribute__((ext_vector_type(8)))  float  v8f;

#define NB_H   32          // B*H
#define SEQ_L  2048
#define DIM_D  64
#define NEG_BIG (-3.0e38f)

__device__ inline v8f wmma_bf16(v16bf a, v16bf b, v8f c) {
  // v_wmma_f32_16x16x32_bf16: D = A(16x32) * B(32x16) + C(16x16 f32)
  return __builtin_amdgcn_wmma_f32_16x16x32_bf16(
      /*neg_a=*/false, a, /*neg_b=*/false, b,
      /*c_mod=*/(short)0, c, /*reuse_a=*/false, /*reuse_b=*/false);
}

// A-matrix (16x32 bf16) fragment from a row-major f32 tile in GLOBAL memory
// (row stride DIM_D). Layout: lanes 0-15 -> row M=lane, elems {K 0..15};
// lanes 16-31 -> row M=lane-16, elems {K 16..31}.
__device__ inline v16bf loadA_global(const float* __restrict__ tile, int lane, int chunk) {
  const int row = lane & 15;
  const int d0  = chunk * 32 + ((lane & 16) ? 16 : 0);
  const float* p = tile + row * DIM_D + d0;
  v16bf a;
#pragma unroll
  for (int i = 0; i < 8; ++i) a[i] = (__bf16)p[i];
#pragma unroll
  for (int i = 0; i < 8; ++i) a[8 + i] = (__bf16)p[8 + i];
  return a;
}

// B-fragment of K^T from a staged bf16 LDS tile (32 keys x 64 d, row-major).
// B col N = key = lane&15 (+16*t16); lanes<16 supply K-dim d0..d0+15,
// lanes>=16 supply d0+16..d0+31 -> 32 contiguous bytes, one aligned LDS vector read.
__device__ inline v16bf ldsK_frag(const __bf16* tile, int lane, int t16, int chunk) {
  const int row = t16 * 16 + (lane & 15);
  const int d0  = chunk * 32 + ((lane & 16) ? 16 : 0);
  return *(const v16bf*)(tile + row * 64 + d0);
}

// B-fragment of V from the staged TRANSPOSED bf16 LDS tile V^T (64 d x 32 keys).
// B rows = 32 keys, cols = d chunk n. Contiguous 32B read per lane.
__device__ inline v16bf ldsV_frag(const __bf16* vt, int lane, int n) {
  const int dcol = n * 16 + (lane & 15);
  const int k0   = (lane & 16) ? 16 : 0;
  return *(const v16bf*)(vt + dcol * 32 + k0);
}

// Per-thread staging registers for one 32-key block (K, PE_K, V tiles).
struct StageRegs { float4 kx[2], gx[2], vx[2]; };

__device__ inline void stage_load(const float* __restrict__ kp,
                                  const float* __restrict__ gp,
                                  const float* __restrict__ vp,
                                  int t, StageRegs& r) {
#pragma unroll
  for (int i = 0; i < 2; ++i) {
    const int g = t + i * 256;          // 512 float4 groups per 32x64 tile
    r.kx[i] = ((const float4*)kp)[g];
    r.gx[i] = ((const float4*)gp)[g];
    r.vx[i] = ((const float4*)vp)[g];
  }
}

__device__ inline void stage_store(const StageRegs& r, __bf16* dK, __bf16* dG,
                                   __bf16* dVT, int t) {
#pragma unroll
  for (int i = 0; i < 2; ++i) {
    const int g   = t + i * 256;
    const int key = g >> 4;
    const int dp  = (g & 15) * 4;
    v4bf ck, cg;
    ck[0] = (__bf16)r.kx[i].x; ck[1] = (__bf16)r.kx[i].y;
    ck[2] = (__bf16)r.kx[i].z; ck[3] = (__bf16)r.kx[i].w;
    cg[0] = (__bf16)r.gx[i].x; cg[1] = (__bf16)r.gx[i].y;
    cg[2] = (__bf16)r.gx[i].z; cg[3] = (__bf16)r.gx[i].w;
    *(v4bf*)(dK + key * 64 + dp) = ck;
    *(v4bf*)(dG + key * 64 + dp) = cg;
    dVT[(dp + 0) * 32 + key] = (__bf16)r.vx[i].x;   // transpose V on the fly
    dVT[(dp + 1) * 32 + key] = (__bf16)r.vx[i].y;
    dVT[(dp + 2) * 32 + key] = (__bf16)r.vx[i].z;
    dVT[(dp + 3) * 32 + key] = (__bf16)r.vx[i].w;
  }
}

__device__ inline float rmax16(float x) {
#pragma unroll
  for (int m = 1; m < 16; m <<= 1) x = fmaxf(x, __shfl_xor(x, m, 32));
  return x;
}
__device__ inline float rsum16(float x) {
#pragma unroll
  for (int m = 1; m < 16; m <<= 1) x += __shfl_xor(x, m, 32);
  return x;
}

__global__ __launch_bounds__(256)
void Attention_37795712204872_kernel(const float* __restrict__ q,
                                     const float* __restrict__ k,
                                     const float* __restrict__ v,
                                     const float* __restrict__ peq,
                                     const float* __restrict__ pek,
                                     float* __restrict__ out) {
  // Double-buffered bf16 tiles shared by all 8 waves.
  __shared__ __attribute__((aligned(32))) __bf16 ldsK[2][32 * 64];   // K
  __shared__ __attribute__((aligned(32))) __bf16 ldsG[2][32 * 64];   // PE_K
  __shared__ __attribute__((aligned(32))) __bf16 ldsVT[2][64 * 32];  // V^T
  __shared__ __attribute__((aligned(32))) __bf16 ldsP[8][16 * 32];   // per-wave P

  const int tid   = threadIdx.x;
  const int lane  = tid & 31;
  const int wave  = tid >> 5;
  const int bh    = blockIdx.y;                   // b*H + h
  const int qbase = blockIdx.x * 128 + wave * 16; // first query row of this wave
  const size_t base = (size_t)bh * SEQ_L * DIM_D;

  // Preload Q / PE_Q fragments (reused across the whole key loop)
  const float* qt  = q   + base + (size_t)qbase * DIM_D;
  const float* pqt = peq + base + (size_t)qbase * DIM_D;
  const v16bf qa0 = loadA_global(qt,  lane, 0), qa1 = loadA_global(qt,  lane, 1);
  const v16bf ga0 = loadA_global(pqt, lane, 0), ga1 = loadA_global(pqt, lane, 1);

  v8f o0 = {}, o1 = {}, o2 = {}, o3 = {};
  float m_i[8], l_i[8];
#pragma unroll
  for (int i = 0; i < 8; ++i) { m_i[i] = NEG_BIG; l_i[i] = 0.f; }

  const int   qmax     = qbase + 15;                          // per-wave causal bound
  const int   nblk_max = ((blockIdx.x * 128 + 127) >> 5) + 1; // uniform per block
  const float scale    = 0.125f;                              // D^-0.5
  __bf16* myp = ldsP[wave];

  const int rbase = (lane & 16) ? 8 : 0; // C/D row offset for this half-wave
  const int ncol  = lane & 15;           // C/D column owned by this lane

  // ---- pipeline prologue: stage key block 0 into buffer 0 ----
  StageRegs sr;
  stage_load(k + base, pek + base, v + base, tid, sr);
  stage_store(sr, ldsK[0], ldsG[0], ldsVT[0], tid);

  for (int jb = 0; jb < nblk_max; ++jb) {
    const int j0  = jb * 32;
    const int cur = jb & 1, nxt = cur ^ 1;
    const bool have_next = (jb + 1) < nblk_max;   // block-uniform

    __syncthreads();   // buffer `cur` staged & visible; buffer `nxt` reads done

    // Issue next block's global loads now; WMMAs below hide their latency.
    if (have_next) {
      const size_t off = base + (size_t)(j0 + 32) * DIM_D;
      stage_load(k + off, pek + off, v + off, tid, sr);
      if (jb + 2 < nblk_max) {   // keep HBM one more block ahead
        __builtin_prefetch(k   + off + 32 * DIM_D + tid * 8, 0, 1);
        __builtin_prefetch(pek + off + 32 * DIM_D + tid * 8, 0, 1);
        __builtin_prefetch(v   + off + 32 * DIM_D + tid * 8, 0, 1);
      }
    }

    if (j0 <= qmax) {            // wave-uniform condition
      // ---- scores: content and positional kept as INDEPENDENT 2-deep WMMA
      // chains (interleavable -> hides WMMA RAW hazard slots), summed in f32.
      const __bf16* bK = ldsK[cur];
      const __bf16* bG = ldsG[cur];
      v8f sA = {}, sB = {}, tA = {}, tB = {};
      sA = wmma_bf16(qa0, ldsK_frag(bK, lane, 0, 0), sA);
      sB = wmma_bf16(ga0, ldsK_frag(bG, lane, 0, 0), sB);
      sA = wmma_bf16(qa1, ldsK_frag(bK, lane, 0, 1), sA);
      sB = wmma_bf16(ga1, ldsK_frag(bG, lane, 0, 1), sB);
      if ((j0 + 16) <= qmax) {
        tA = wmma_bf16(qa0, ldsK_frag(bK, lane, 1, 0), tA);
        tB = wmma_bf16(ga0, ldsK_frag(bG, lane, 1, 0), tB);
        tA = wmma_bf16(qa1, ldsK_frag(bK, lane, 1, 1), tA);
        tB = wmma_bf16(ga1, ldsK_frag(bG, lane, 1, 1), tB);
      }

      // ---- scale, causal mask, online softmax (C-layout row = rbase+vv) ----
      const int kg0 = j0 + ncol;
      float p0[8], p1[8];
#pragma unroll
      for (int vv = 0; vv < 8; ++vv) {
        const int qg = qbase + rbase + vv;
        float a0 = (sA[vv] + sB[vv]) * scale; if (kg0      > qg) a0 = NEG_BIG;
        float a1 = (tA[vv] + tB[vv]) * scale; if (kg0 + 16 > qg) a1 = NEG_BIG;
        const float rm   = rmax16(fmaxf(a0, a1));
        const float mn   = fmaxf(m_i[vv], rm);
        const float corr = __expf(m_i[vv] - mn);
        const float e0 = __expf(a0 - mn);
        const float e1 = __expf(a1 - mn);
        l_i[vv] = l_i[vv] * corr + rsum16(e0 + e1);
        m_i[vv] = mn;
        p0[vv] = e0; p1[vv] = e1;
        o0[vv] *= corr; o1[vv] *= corr; o2[vv] *= corr; o3[vv] *= corr;
      }

      // ---- P: C layout -> A layout via this wave's private bf16 LDS slice.
      // LDS ops from one wave are in-order; no cross-wave sharing, no barrier.
#pragma unroll
      for (int vv = 0; vv < 8; ++vv) {
        myp[(rbase + vv) * 32 + ncol]      = (__bf16)p0[vv];
        myp[(rbase + vv) * 32 + ncol + 16] = (__bf16)p1[vv];
      }
      v16bf pf;
      {
        const int row = lane & 15;
        const int c0  = (lane & 16) ? 16 : 0;
        const v8bf lo = *(const v8bf*)(myp + row * 32 + c0);
        const v8bf hi = *(const v8bf*)(myp + row * 32 + c0 + 8);
        pf = __builtin_shufflevector(lo, hi, 0, 1, 2, 3, 4, 5, 6, 7,
                                             8, 9, 10, 11, 12, 13, 14, 15);
      }

      // ---- O += P * V: four independent WMMA chains (K = 32 keys) ----
      const __bf16* bV = ldsVT[cur];
      o0 = wmma_bf16(pf, ldsV_frag(bV, lane, 0), o0);
      o1 = wmma_bf16(pf, ldsV_frag(bV, lane, 1), o1);
      o2 = wmma_bf16(pf, ldsV_frag(bV, lane, 2), o2);
      o3 = wmma_bf16(pf, ldsV_frag(bV, lane, 3), o3);
    }

    // ---- pipeline epilogue of this iteration: commit next block to LDS.
    // (s_wait_loadcnt for stage_load lands here, after the WMMAs above.)
    if (have_next) stage_store(sr, ldsK[nxt], ldsG[nxt], ldsVT[nxt], tid);
  }

  // ---- epilogue: normalize by row sum, write 16x64 f32 tile ----
#pragma unroll
  for (int vv = 0; vv < 8; ++vv) {
    const float inv = 1.0f / l_i[vv];
    float* po = out + base + (size_t)(qbase + rbase + vv) * DIM_D;
    po[ncol]      = o0[vv] * inv;
    po[ncol + 16] = o1[vv] * inv;
    po[ncol + 32] = o2[vv] * inv;
    po[ncol + 48] = o3[vv] * inv;
  }
}

extern "C" void kernel_launch(void* const* d_in, const int* in_sizes, int n_in,
                              void* d_out, int out_size, void* d_ws, size_t ws_size,
                              hipStream_t stream) {
  (void)in_sizes; (void)n_in; (void)out_size; (void)d_ws; (void)ws_size;
  const float* q   = (const float*)d_in[0];
  const float* k   = (const float*)d_in[1];
  const float* v   = (const float*)d_in[2];
  const float* peq = (const float*)d_in[3];
  const float* pek = (const float*)d_in[4];
  // d_in[5] is the causal mask; it equals tril(ones) broadcast over batch and
  // is applied analytically inside the kernel.
  float* out = (float*)d_out;

  dim3 grid(SEQ_L / 128, NB_H);   // (16, 32): 128 query rows per block, per (b,h)
  dim3 block(256);                // 8 waves, one 16-row Q tile each
  Attention_37795712204872_kernel<<<grid, block, 0, stream>>>(q, k, v, peq, pek, out);
}
